// WeightOnlyQuantLinear_28630251995563
// MI455X (gfx1250) — compile-verified, matrix operand order checked by reading, not run on hardware
//
#include <hip/hip_runtime.h>

typedef __attribute__((ext_vector_type(16))) _Float16 v16h;
typedef __attribute__((ext_vector_type(8)))  _Float16 v8h;
typedef __attribute__((ext_vector_type(4)))  _Float16 v4h;
typedef __attribute__((ext_vector_type(2)))  _Float16 v2h;
typedef __attribute__((ext_vector_type(8)))  float    v8f;

namespace {
constexpr int Mtot = 8192;    // B*S = 4*2048
constexpr int Ntot = 4096;    // OUT
constexpr int Ktot = 4096;    // IN
constexpr int BM = 128, BN = 256, BK = 32;
constexpr int LDA = 40;       // f16 row stride in LDS: 80 bytes (16B-aligned rows, 20-dword stride)
constexpr int KB2 = Ktot / 2; // packed-byte (int32 element) count per output row
}

union FragU { v16h v; v8h h[2]; };

// int4 dequant via f16 magic number: (nib | 0x6400) == f16(1024 + nib);
// one packed subtract of 1032.0 gives {lo-8, hi-8} with zero cvt instructions.
__device__ __forceinline__ unsigned dq_pair(int p) {
    union { unsigned u; v2h h; } t;
    t.u = (unsigned)((p & 15) | ((p & 0xF0) << 12)) | 0x64006400u;
    v2h c = {(_Float16)1032.0f, (_Float16)1032.0f};
    t.h = t.h - c;
    return t.u;
}

__global__ __launch_bounds__(256)
void w4a16_wmma_gemm(const float* __restrict__ x,
                     const int*   __restrict__ qw,
                     const float* __restrict__ scale_p,
                     const float* __restrict__ bias,
                     float* __restrict__ out)
{
    __shared__ _Float16 sA[BM * LDA];  // 10240 B
    __shared__ _Float16 sB[BN * LDA];  // 20480 B

    const int tid    = threadIdx.x;
    const int lane   = tid & 31;
    const int wave   = tid >> 5;       // 0..7
    const int waveM  = wave >> 2;      // 0..1  -> 64 rows each
    const int waveN  = wave & 3;       // 0..3  -> 64 cols each
    const int l16    = lane & 15;
    const int hiHalf = lane >> 4;      // 0: lanes 0-15, 1: lanes 16-31

    const int m0 = blockIdx.y * BM;
    const int n0 = blockIdx.x * BN;

    v8f acc[4][4];
    #pragma unroll
    for (int mi = 0; mi < 4; ++mi)
        #pragma unroll
        for (int ni = 0; ni < 4; ++ni) {
            v8f z = {0.f, 0.f, 0.f, 0.f, 0.f, 0.f, 0.f, 0.f};
            acc[mi][ni] = z;
        }

    for (int kt = 0; kt < Ktot / BK; ++kt) {
        const int k0 = kt * BK;

        // ---- stage A: x[m0..+127][k0..+31] fp32 -> f16 LDS (4 float4 / thread) ----
        #pragma unroll
        for (int j = 0; j < 4; ++j) {
            int idx = tid + j * 256;          // 1024 float4 slots = 128 rows x 8
            int row = idx >> 3;
            int c4  = idx & 7;
            const float4 f = *reinterpret_cast<const float4*>(
                x + (size_t)(m0 + row) * Ktot + k0 + c4 * 4);
            v4h h = {(_Float16)f.x, (_Float16)f.y, (_Float16)f.z, (_Float16)f.w};
            *reinterpret_cast<v4h*>(&sA[row * LDA + c4 * 4]) = h;
        }

        // ---- stage B: packed int4 -> f16 LDS (4 int4 / thread, magic-number dequant) ----
        #pragma unroll
        for (int j = 0; j < 4; ++j) {
            int idx = tid + j * 256;          // 1024 int4 slots = 256 rows x 4
            int row = idx >> 2;
            int q   = idx & 3;
            const int4 p = *reinterpret_cast<const int4*>(
                qw + (size_t)(n0 + row) * KB2 + (k0 >> 1) + q * 4);
            uint4 h;
            h.x = dq_pair(p.x);
            h.y = dq_pair(p.y);
            h.z = dq_pair(p.z);
            h.w = dq_pair(p.w);
            *reinterpret_cast<uint4*>(&sB[row * LDA + q * 8]) = h;
        }

        // prefetch next K-slice of activations
        if (kt + 1 < Ktot / BK)
            __builtin_prefetch(x + (size_t)(m0 + (tid >> 1)) * Ktot + k0 + BK, 0, 0);

        __syncthreads();

        // ---- fragments: CDNA5 16-bit A layout (lanes 0-15: K0..7/16..23; 16-31: K8..15/24..31) ----
        FragU aF[4], bF[4];
        #pragma unroll
        for (int mi = 0; mi < 4; ++mi) {
            int row = waveM * 64 + mi * 16 + l16;
            const _Float16* s = &sA[row * LDA + hiHalf * 8];
            aF[mi].h[0] = *reinterpret_cast<const v8h*>(s);
            aF[mi].h[1] = *reinterpret_cast<const v8h*>(s + 16);
        }
        #pragma unroll
        for (int ni = 0; ni < 4; ++ni) {
            int row = waveN * 64 + ni * 16 + l16;
            const _Float16* s = &sB[row * LDA + hiHalf * 8];
            bF[ni].h[0] = *reinterpret_cast<const v8h*>(s);
            bF[ni].h[1] = *reinterpret_cast<const v8h*>(s + 16);
        }

        // ---- 16 WMMAs per wave per K-step (1.0 ds_load_b128 per WMMA) ----
        #pragma unroll
        for (int mi = 0; mi < 4; ++mi)
            #pragma unroll
            for (int ni = 0; ni < 4; ++ni)
                acc[mi][ni] = __builtin_amdgcn_wmma_f32_16x16x32_f16(
                    false, aF[mi].v, false, bF[ni].v,
                    (short)0, acc[mi][ni], false, false);

        __syncthreads();
    }

    // ---- epilogue: y = acc*scale + bias (C layout: vgpr r -> M=r / r+8, lane -> N) ----
    const float s = *scale_p;
    #pragma unroll
    for (int ni = 0; ni < 4; ++ni) {
        int n   = n0 + waveN * 64 + ni * 16 + l16;
        float b = bias[n];
        #pragma unroll
        for (int mi = 0; mi < 4; ++mi) {
            int mbase = m0 + waveM * 64 + mi * 16 + hiHalf * 8;
            #pragma unroll
            for (int r = 0; r < 8; ++r)
                out[(size_t)(mbase + r) * Ntot + n] = acc[mi][ni][r] * s + b;
        }
    }
}

extern "C" void kernel_launch(void* const* d_in, const int* in_sizes, int n_in,
                              void* d_out, int out_size, void* d_ws, size_t ws_size,
                              hipStream_t stream) {
    const float* x    = (const float*)d_in[0];
    const int*   qw   = (const int*)d_in[1];
    const float* sc   = (const float*)d_in[2];
    const float* bias = (const float*)d_in[3];
    float* out = (float*)d_out;

    dim3 grid(Ntot / BN, Mtot / BM);   // (16, 64)
    w4a16_wmma_gemm<<<grid, 256, 0, stream>>>(x, qw, sc, bias, out);
}
